// LowRankAttention_12137577579271
// MI455X (gfx1250) — compile-verified, matrix operand order checked by reading, not run on hardware
//
#include <hip/hip_runtime.h>
#include <hip/hip_bf16.h>

// ---------------------------------------------------------------------------
// LowRankAttention for MI455X (gfx1250), wave32 + bf16 WMMA, fp32-split bf16.
//   out[:,0:100]   = (U @ (V^T Z)) * Dscale
//   out[:,100:200] = T
// where [U|V|Z|T] = relu(X @ W^T + b)
// All f32->bf16(hi,lo) splitting is hoisted out of the WMMA hot loops so the
// inner loops are pure global_load_b128 + v_wmma_f32_16x16x32_bf16.
// ---------------------------------------------------------------------------

#define N_ROWS 100000
#define D_DIM  512
#define M_COLS 400          // 4*K
#define KC     100          // K
#define KPAD   128          // K padded for U / VtZ fragments
#define MPAD   112          // 100 padded to 7 tiles of 16
#define OUTC   200
#define WPB    8            // waves per block (256 threads)
#define ROWG   5            // row tiles per wave in main GEMM (6250 = 5*1250)
#define VTZ_SPLITS 64

typedef __bf16 bf16_t;
typedef bf16_t       v16bf __attribute__((ext_vector_type(16)));
typedef float        v8f   __attribute__((ext_vector_type(8)));
typedef unsigned int u32x4 __attribute__((ext_vector_type(4)));
typedef float        f32x4 __attribute__((ext_vector_type(4)));

// 32-byte A/B fragment viewed either as bf16 lanes or as two 16B load chunks
union Frag16 { v16bf v; u32x4 q[2]; };

__device__ __forceinline__ void split_bf16(float x, bf16_t& hi, bf16_t& lo) {
    hi = (bf16_t)x;                 // RNE f32 -> bf16
    lo = (bf16_t)(x - (float)hi);   // exact residual, re-rounded
}

// ---------------------------------------------------------------------------
// Split a f32 array into bf16 hi/lo arrays (n multiple of 4).
// ---------------------------------------------------------------------------
__global__ void split_kernel(const float* __restrict__ src,
                             bf16_t* __restrict__ hi, bf16_t* __restrict__ lo, int n)
{
    const int i = (blockIdx.x * blockDim.x + threadIdx.x) * 4;
    if (i >= n) return;
    const f32x4 v = *(const f32x4*)(src + i);
#pragma unroll
    for (int j = 0; j < 4; ++j) {
        bf16_t h, l;
        split_bf16(v[j], h, l);
        hi[i + j] = h; lo[i + j] = l;
    }
}

// ---------------------------------------------------------------------------
// Grid-stride 16B zero fill.
// ---------------------------------------------------------------------------
__global__ void zero4_kernel(u32x4* __restrict__ p, long n4)
{
    const u32x4 z = {0u, 0u, 0u, 0u};
    for (long i = blockIdx.x * (long)blockDim.x + threadIdx.x; i < n4;
         i += (long)gridDim.x * blockDim.x)
        p[i] = z;
}

// ---------------------------------------------------------------------------
// Kernel 1: tmp = relu(X @ W^T + b).
// One wave = 5 row-tiles (80 rows) x 1 col-tile (16 cols).
// PRE=true: A fragments are direct b128 loads from pre-split Xhi/Xlo.
// Epilogue: U -> bf16 hi/lo row-major [N][128]; V,Z -> bf16 hi/lo transposed
// [col][N]; T -> out (f32); U/V column sums via one atomicAdd per lane.
// ---------------------------------------------------------------------------
template <bool PRE>
__global__ void __launch_bounds__(256)
gemm_relu_kernel(const float* __restrict__ X,
                 const bf16_t* __restrict__ Xhi, const bf16_t* __restrict__ Xlo,
                 const bf16_t* __restrict__ Whi, const bf16_t* __restrict__ Wlo,
                 const float* __restrict__ bias,
                 bf16_t* __restrict__ Uhi, bf16_t* __restrict__ Ulo,
                 bf16_t* __restrict__ Vthi, bf16_t* __restrict__ Vtlo,
                 bf16_t* __restrict__ Zthi, bf16_t* __restrict__ Ztlo,
                 float* __restrict__ sums, float* __restrict__ out)
{
    const int lane  = threadIdx.x & 31;
    const int gwave = blockIdx.x * WPB + (threadIdx.x >> 5);
    const int totalWaves = ((N_ROWS / 16) / ROWG) * (M_COLS / 16);  // 31250
    if (gwave >= totalWaves) return;                                // wave-uniform

    const int colTile  = gwave % (M_COLS / 16);
    const int rowGroup = gwave / (M_COLS / 16);
    const int colBase  = colTile << 4;
    const int half     = lane >> 4;
    const int lidx     = lane & 15;

    const int wofs = (colBase + lidx) * D_DIM;   // B fragment source row of W

    v8f acc[ROWG];
#pragma unroll
    for (int t = 0; t < ROWG; ++t) acc[t] = {};

    for (int kt = 0; kt < D_DIM / 32; ++kt) {
        const int kb = kt << 5;
        const int o0 = kb + (half << 3);         // k = kb + half*8 .. +8
        const int o1 = kb + 16 + (half << 3);    // k = kb+16+half*8 .. +8

        Frag16 bhi, blo;
        bhi.q[0] = *(const u32x4*)(Whi + wofs + o0);
        bhi.q[1] = *(const u32x4*)(Whi + wofs + o1);
        blo.q[0] = *(const u32x4*)(Wlo + wofs + o0);
        blo.q[1] = *(const u32x4*)(Wlo + wofs + o1);

#pragma unroll
        for (int t = 0; t < ROWG; ++t) {
            const int row = ((rowGroup * ROWG + t) << 4) + lidx;
            Frag16 ahi, alo;
            if (PRE) {
                const bf16_t* xh = Xhi + row * D_DIM;
                const bf16_t* xl = Xlo + row * D_DIM;
                __builtin_prefetch(xh + kb + 32, 0, 1);  // global_prefetch_b8
                ahi.q[0] = *(const u32x4*)(xh + o0);
                ahi.q[1] = *(const u32x4*)(xh + o1);
                alo.q[0] = *(const u32x4*)(xl + o0);
                alo.q[1] = *(const u32x4*)(xl + o1);
            } else {
                const float* xrow = X + row * D_DIM;
                __builtin_prefetch(xrow + kb + 32, 0, 1);
#pragma unroll
                for (int p = 0; p < 8; ++p) {
                    const int k = kb + ((p >> 2) << 4) + (half << 3) + ((p & 3) << 1);
                    bf16_t h0, l0, h1, l1;
                    split_bf16(xrow[k],     h0, l0);
                    split_bf16(xrow[k + 1], h1, l1);
                    ahi.v[2*p] = h0; alo.v[2*p] = l0;
                    ahi.v[2*p+1] = h1; alo.v[2*p+1] = l1;
                }
            }
            acc[t] = __builtin_amdgcn_wmma_f32_16x16x32_bf16(false, alo.v, false, bhi.v, (short)0, acc[t], false, false);
            acc[t] = __builtin_amdgcn_wmma_f32_16x16x32_bf16(false, ahi.v, false, blo.v, (short)0, acc[t], false, false);
            acc[t] = __builtin_amdgcn_wmma_f32_16x16x32_bf16(false, ahi.v, false, bhi.v, (short)0, acc[t], false, false);
        }
    }

    // ---- epilogue: bias + relu; route column; fold in column sums ----
    const int col = colBase + lidx;
    const float bv = bias[col];
    bf16_t *ph = nullptr, *pl = nullptr; float* pT = nullptr;
    long A; int S;
    if (col < KC)          { ph = Uhi;  pl = Ulo;  A = col;                        S = KPAD; }
    else if (col < 2 * KC) { ph = Vthi; pl = Vtlo; A = (long)(col - KC) * N_ROWS;  S = 1;    }
    else if (col < 3 * KC) { ph = Zthi; pl = Ztlo; A = (long)(col - 2*KC) * N_ROWS; S = 1;   }
    else                   { pT = out;             A = KC + (col - 3 * KC);        S = OUTC; }

    float s = 0.f;
#pragma unroll
    for (int t = 0; t < ROWG; ++t) {
        const int row0 = ((rowGroup * ROWG + t) << 4) + (half << 3);
#pragma unroll
        for (int r = 0; r < 8; ++r) {
            float v = acc[t][r] + bv;
            v = v > 0.f ? v : 0.f;
            const long idx = A + (long)(row0 + r) * S;
            if (pT) {
                pT[idx] = v;
            } else {
                bf16_t h, l;
                split_bf16(v, h, l);
                ph[idx] = h; pl[idx] = l;
                s += v;
            }
        }
    }
    if (col < 2 * KC) atomicAdd(&sums[col], s);   // [0..99]=U sums, [100..199]=V sums
}

// ---------------------------------------------------------------------------
// Dscale = 1 / (dot(sumU,sumV)/N + 1e-6)
// ---------------------------------------------------------------------------
__global__ void dscale_kernel(const float* __restrict__ sums, float* __restrict__ dscale)
{
    __shared__ float red[128];
    const int t = threadIdx.x;
    red[t] = (t < KC) ? sums[t] * sums[KC + t] : 0.f;
    __syncthreads();
    for (int s = 64; s > 0; s >>= 1) {
        if (t < s) red[t] += red[t + s];
        __syncthreads();
    }
    if (t == 0) dscale[0] = 1.0f / (red[0] / (float)N_ROWS + 1e-6f);
}

// ---------------------------------------------------------------------------
// VtZ = V^T @ Z (100x100, reduction over N). Split-K WMMA + f32 atomics.
// A/B fragments are contiguous b128 loads from transposed bf16 hi/lo arrays.
// Pad rows (>=100) hold garbage but only affect guarded (unstored) outputs.
// ---------------------------------------------------------------------------
__global__ void __launch_bounds__(256)
vtz_kernel(const bf16_t* __restrict__ Vthi, const bf16_t* __restrict__ Vtlo,
           const bf16_t* __restrict__ Zthi, const bf16_t* __restrict__ Ztlo,
           float* __restrict__ VtZ)
{
    const int lane  = threadIdx.x & 31;
    const int gwave = blockIdx.x * WPB + (threadIdx.x >> 5);
    const int TILES = 7 * 7;
    if (gwave >= TILES * VTZ_SPLITS) return;     // wave-uniform
    const int tile  = gwave % TILES;
    const int split = gwave / TILES;
    const int mBase = (tile / 7) << 4, nBase = (tile % 7) << 4;
    const int half  = lane >> 4, lidx = lane & 15;
    const int mcol  = mBase + lidx, ncol = nBase + lidx;   // <= 111 < MPAD

    const bf16_t* ah = Vthi + (size_t)mcol * N_ROWS;
    const bf16_t* al = Vtlo + (size_t)mcol * N_ROWS;
    const bf16_t* bh = Zthi + (size_t)ncol * N_ROWS;
    const bf16_t* bl = Ztlo + (size_t)ncol * N_ROWS;

    v8f acc = {};
    for (int kt = split; kt < N_ROWS / 32; kt += VTZ_SPLITS) {
        const int kb = kt << 5;
        const int o0 = kb + (half << 3);
        const int o1 = kb + 16 + (half << 3);
        Frag16 ahi, alo, bhi, blo;
        ahi.q[0] = *(const u32x4*)(ah + o0); ahi.q[1] = *(const u32x4*)(ah + o1);
        alo.q[0] = *(const u32x4*)(al + o0); alo.q[1] = *(const u32x4*)(al + o1);
        bhi.q[0] = *(const u32x4*)(bh + o0); bhi.q[1] = *(const u32x4*)(bh + o1);
        blo.q[0] = *(const u32x4*)(bl + o0); blo.q[1] = *(const u32x4*)(bl + o1);
        acc = __builtin_amdgcn_wmma_f32_16x16x32_bf16(false, alo.v, false, bhi.v, (short)0, acc, false, false);
        acc = __builtin_amdgcn_wmma_f32_16x16x32_bf16(false, ahi.v, false, blo.v, (short)0, acc, false, false);
        acc = __builtin_amdgcn_wmma_f32_16x16x32_bf16(false, ahi.v, false, bhi.v, (short)0, acc, false, false);
    }
    if (ncol < KC) {
#pragma unroll
        for (int r = 0; r < 8; ++r) {
            const int m = mBase + (half << 3) + r;
            if (m < KC) atomicAdd(&VtZ[m * KC + ncol], acc[r]);
        }
    }
}

// ---------------------------------------------------------------------------
// Transpose + split + zero-pad VtZ (100x100 f32) -> VtZt hi/lo [112][128] bf16
// so res_kernel's B fragments are clean b128 loads (k-padding exactly zero).
// ---------------------------------------------------------------------------
__global__ void prep_vtz_kernel(const float* __restrict__ VtZ,
                                bf16_t* __restrict__ th, bf16_t* __restrict__ tl)
{
    const int i = blockIdx.x * blockDim.x + threadIdx.x;   // over MPAD*KPAD
    if (i >= MPAD * KPAD) return;
    const int n = i >> 7, k = i & (KPAD - 1);
    const float v = (n < KC && k < KC) ? VtZ[k * KC + n] : 0.f;
    bf16_t h, l;
    split_bf16(v, h, l);
    th[i] = h; tl[i] = l;
}

// ---------------------------------------------------------------------------
// res = (U @ VtZ) * Dscale -> out[:, 0:100]. K padded 100 -> 128 (zeros).
// ---------------------------------------------------------------------------
__global__ void __launch_bounds__(256)
res_kernel(const bf16_t* __restrict__ Uhi, const bf16_t* __restrict__ Ulo,
           const bf16_t* __restrict__ Bth, const bf16_t* __restrict__ Btl,
           const float* __restrict__ dscale, float* __restrict__ out)
{
    const int lane  = threadIdx.x & 31;
    const int gwave = blockIdx.x * WPB + (threadIdx.x >> 5);
    const int rowTiles = N_ROWS / 16;       // 6250
    if (gwave >= rowTiles * 7) return;      // wave-uniform
    const int colTile = gwave % 7;
    const int rowTile = gwave / 7;
    const int half = lane >> 4, lidx = lane & 15;
    const int row  = (rowTile << 4) + lidx;
    const int ncol = (colTile << 4) + lidx;   // <= 111 < MPAD

    const bf16_t* ah = Uhi + (size_t)row * KPAD;
    const bf16_t* al = Ulo + (size_t)row * KPAD;
    const bf16_t* bh = Bth + (size_t)ncol * KPAD;
    const bf16_t* bl = Btl + (size_t)ncol * KPAD;

    v8f acc = {};
#pragma unroll
    for (int kt = 0; kt < KPAD / 32; ++kt) {
        const int kb = kt << 5;
        const int o0 = kb + (half << 3);
        const int o1 = kb + 16 + (half << 3);
        Frag16 ahi, alo, bhi, blo;
        ahi.q[0] = *(const u32x4*)(ah + o0); ahi.q[1] = *(const u32x4*)(ah + o1);
        alo.q[0] = *(const u32x4*)(al + o0); alo.q[1] = *(const u32x4*)(al + o1);
        bhi.q[0] = *(const u32x4*)(bh + o0); bhi.q[1] = *(const u32x4*)(bh + o1);
        blo.q[0] = *(const u32x4*)(bl + o0); blo.q[1] = *(const u32x4*)(bl + o1);
        acc = __builtin_amdgcn_wmma_f32_16x16x32_bf16(false, alo.v, false, bhi.v, (short)0, acc, false, false);
        acc = __builtin_amdgcn_wmma_f32_16x16x32_bf16(false, ahi.v, false, blo.v, (short)0, acc, false, false);
        acc = __builtin_amdgcn_wmma_f32_16x16x32_bf16(false, ahi.v, false, bhi.v, (short)0, acc, false, false);
    }
    const float ds = dscale[0];
    if (ncol < KC) {
#pragma unroll
        for (int r = 0; r < 8; ++r) {
            const int m = (rowTile << 4) + (half << 3) + r;
            out[(long)m * OUTC + ncol] = acc[r] * ds;
        }
    }
}

// ---------------------------------------------------------------------------
// Launch
// ---------------------------------------------------------------------------
extern "C" void kernel_launch(void* const* d_in, const int* in_sizes, int n_in,
                              void* d_out, int out_size, void* d_ws, size_t ws_size,
                              hipStream_t stream)
{
    (void)in_sizes; (void)n_in; (void)out_size;
    const float* X  = (const float*)d_in[0];   // 100000 x 512
    const float* W  = (const float*)d_in[1];   // 400 x 512
    const float* bb = (const float*)d_in[2];   // 400
    float* out = (float*)d_out;                // 100000 x 200

    // ---- workspace carve-up (all regions 64B aligned) ----
    char* base = (char*)d_ws;
    size_t off = 0;
    auto carve = [&](size_t bytes) -> char* {
        char* p = base + off;
        off += (bytes + 63) & ~(size_t)63;
        return p;
    };
    bf16_t* Uhi  = (bf16_t*)carve((size_t)N_ROWS * KPAD * 2);
    bf16_t* Ulo  = (bf16_t*)carve((size_t)N_ROWS * KPAD * 2);   // contiguous w/ Uhi
    bf16_t* Vthi = (bf16_t*)carve((size_t)MPAD * N_ROWS * 2);
    bf16_t* Vtlo = (bf16_t*)carve((size_t)MPAD * N_ROWS * 2);
    bf16_t* Zthi = (bf16_t*)carve((size_t)MPAD * N_ROWS * 2);
    bf16_t* Ztlo = (bf16_t*)carve((size_t)MPAD * N_ROWS * 2);
    bf16_t* Whi  = (bf16_t*)carve((size_t)M_COLS * D_DIM * 2);
    bf16_t* Wlo  = (bf16_t*)carve((size_t)M_COLS * D_DIM * 2);
    float*  sums   = (float*)carve(256 * 4);                    // [U(100)|V(100)|pad]
    float*  VtZ    = (float*)carve(KC * KC * 4);                // contiguous w/ sums
    bf16_t* VtZthi = (bf16_t*)carve((size_t)MPAD * KPAD * 2);
    bf16_t* VtZtlo = (bf16_t*)carve((size_t)MPAD * KPAD * 2);
    float*  dscale = (float*)carve(64);
    const size_t baseBytes = off;
    bf16_t* Xhi = (bf16_t*)carve((size_t)N_ROWS * D_DIM * 2);
    bf16_t* Xlo = (bf16_t*)carve((size_t)N_ROWS * D_DIM * 2);
    const size_t fullBytes = off;
    const bool preX = (ws_size >= fullBytes);   // ws_size fixed -> deterministic
    (void)baseBytes;

    // 1. zero: U hi/lo (k-padding must be 0) and sums+VtZ (atomics targets)
    zero4_kernel<<<2048, 256, 0, stream>>>((u32x4*)Uhi,
                                           (long)((size_t)N_ROWS * KPAD * 2 * 2 / 16));
    zero4_kernel<<<16, 256, 0, stream>>>((u32x4*)sums,
                                         (long)((256 * 4 + KC * KC * 4) / 16));
    // 2. pre-split W (always) and X (if ws allows)
    split_kernel<<<(M_COLS * D_DIM / 4 + 255) / 256, 256, 0, stream>>>(W, Whi, Wlo, M_COLS * D_DIM);
    if (preX)
        split_kernel<<<(N_ROWS * D_DIM / 4 + 255) / 256, 256, 0, stream>>>(X, Xhi, Xlo, N_ROWS * D_DIM);
    // 3. fused GEMM + bias + relu + scatter + column sums
    {
        const int waves  = ((N_ROWS / 16) / ROWG) * (M_COLS / 16);  // 31250
        const int blocks = (waves + WPB - 1) / WPB;
        if (preX)
            gemm_relu_kernel<true><<<blocks, 256, 0, stream>>>(X, Xhi, Xlo, Whi, Wlo, bb,
                Uhi, Ulo, Vthi, Vtlo, Zthi, Ztlo, sums, out);
        else
            gemm_relu_kernel<false><<<blocks, 256, 0, stream>>>(X, Xhi, Xlo, Whi, Wlo, bb,
                Uhi, Ulo, Vthi, Vtlo, Zthi, Ztlo, sums, out);
    }
    // 4. Dscale
    dscale_kernel<<<1, 128, 0, stream>>>(sums, dscale);
    // 5. VtZ = V^T @ Z (split-K)
    {
        const int waves  = 49 * VTZ_SPLITS;
        const int blocks = (waves + WPB - 1) / WPB;
        vtz_kernel<<<blocks, 256, 0, stream>>>(Vthi, Vtlo, Zthi, Ztlo, VtZ);
    }
    // 6. transpose/split/pad VtZ for the final GEMM's B operand
    prep_vtz_kernel<<<(MPAD * KPAD + 255) / 256, 256, 0, stream>>>(VtZ, VtZthi, VtZtlo);
    // 7. res = U @ VtZ * Dscale -> out[:, 0:100]
    {
        const int waves  = (N_ROWS / 16) * 7;
        const int blocks = (waves + WPB - 1) / WPB;
        res_kernel<<<blocks, 256, 0, stream>>>(Uhi, Ulo, VtZthi, VtZtlo, dscale, out);
    }
}
